// RNNModel_75823352643811
// MI455X (gfx1250) — matrix operand order repaired and checked by reading, not compiled, not run on hardware
//
#include <hip/hip_runtime.h>
#include <hip/hip_bf16.h>
#include <math.h>

// ---------------------------------------------------------------------------
// Problem constants (match reference)
// ---------------------------------------------------------------------------
#define BB   1024
#define SS   128
#define VV   50257
#define DEMB 5
#define H1   150
#define H2   100
#define H1P  160          // H1 padded to multiple of 32 (WMMA K) / 16 (N tiles)
#define H2NP 112          // H2 padded to multiple of 16 for N tiling
#define H2KP 128          // H2 padded to multiple of 32 for WMMA K
#define NT   ((VV + 15) / 16)   // 3142 N-tiles in dense layer
#define NDP  (NT * 16)          // 50272 padded vocab rows for transposed Wd

#define NK1  (H1P / 32)   // 5 K-steps over H1
#define NK2  (H2KP / 32)  // 4 K-steps over H2

// ---------------------------------------------------------------------------
// WMMA fragment types (gfx1250, wave32). bf16 16x16x32: A/B = v16bf, C/D = v8f
// ---------------------------------------------------------------------------
typedef __attribute__((ext_vector_type(16))) __bf16       v16bf;
typedef __attribute__((ext_vector_type(8)))  float        v8f;
typedef __attribute__((ext_vector_type(4)))  unsigned int v4u;

union ABFrag { v16bf v; v4u q[2]; };

static __device__ __forceinline__ unsigned short f2bf(float f) {
  unsigned int u = __float_as_uint(f);
  u += 0x7FFFu + ((u >> 16) & 1u);       // round-to-nearest-even
  return (unsigned short)(u >> 16);
}

// gfx1250 hardware tanh (TRANS32 op) -- one v_tanh_f32 instead of branchy libm.
static __device__ __forceinline__ float fast_tanh(float x) {
#if __has_builtin(__builtin_amdgcn_tanhf)
  return __builtin_amdgcn_tanhf(x);
#else
  float r;
  asm("v_tanh_f32 %0, %1" : "=v"(r) : "v"(x));
  return r;
#endif
}

// Monotonic float<->uint mapping so we can use integer atomicMax for row max.
static __device__ __forceinline__ unsigned int f32_ordered(float f) {
  unsigned int u = __float_as_uint(f);
  return (u & 0x80000000u) ? ~u : (u | 0x80000000u);
}
static __device__ __forceinline__ float f32_unordered(unsigned int u) {
  unsigned int b = (u & 0x80000000u) ? (u & 0x7FFFFFFFu) : ~u;
  return __uint_as_float(b);
}

// A fragment (16x32 bf16, M x K): lane holds row M=lane%16.
// Lanes 0-15: K = kbase+0..7 (q[0]) and kbase+16..23 (q[1]);
// lanes 16-31: K = kbase+8..15 and kbase+24..31.  (ISA 7.12.2 table)
static __device__ __forceinline__ v16bf load_a_frag(const unsigned short* base,
                                                    int ld, int kbase, int lane) {
  const int m  = lane & 15;
  const int kh = (lane < 16) ? 0 : 8;
  const unsigned short* p = base + (size_t)m * ld + kbase + kh;
  ABFrag a;
  a.q[0] = *(const v4u*)(p);
  a.q[1] = *(const v4u*)(p + 16);
  return a.v;
}

// B fragment (32x16 bf16, K x N) from K-major (transposed) weights wt[N][ld]:
// lane holds column N = n0 + lane%16; lanes 0-15 K=kbase+0..7,16..23,
// lanes 16-31 K=kbase+8..15,24..31 (same order as A frag).
static __device__ __forceinline__ v16bf load_b_frag(const unsigned short* wt,
                                                    int ld, int n0, int kbase, int lane) {
  const int n  = n0 + (lane & 15);
  const int ko = (lane < 16) ? 0 : 8;
  const unsigned short* p = wt + (size_t)n * ld + kbase + ko;
  ABFrag b;
  b.q[0] = *(const v4u*)(p);
  b.q[1] = *(const v4u*)(p + 16);
  return b.v;
}

#define WMMA_BF16(a, b, c) \
  __builtin_amdgcn_wmma_f32_16x16x32_bf16(false, (a), false, (b), (short)0, (c), false, false)

// ---------------------------------------------------------------------------
// Kernel 0: clear softmax row stats
// ---------------------------------------------------------------------------
__global__ void init_stats_kernel(unsigned int* rmax, float* rsum) {
  int i = blockIdx.x * blockDim.x + threadIdx.x;
  if (i < BB) { rmax[i] = 0u; rsum[i] = 0.f; }
}

// ---------------------------------------------------------------------------
// Kernel 1: embedding gather + RNN1 input projection, xw1[b,s,:] = emb[tok]@W1 + b1
// (K=5: pure VALU, bandwidth-trivial). Padded cols [150,160) set to 0.
// ---------------------------------------------------------------------------
__global__ void embed_proj_kernel(const int* tokens, const float* emb,
                                  const float* W1, const float* b1, float* xw1) {
  size_t idx = (size_t)blockIdx.x * blockDim.x + threadIdx.x;
  if (idx >= (size_t)BB * SS * H1P) return;
  int    h  = (int)(idx % H1P);
  size_t bs = idx / H1P;
  float acc = 0.f;
  if (h < H1) {
    int tok = tokens[bs];
    acc = b1[h];
#pragma unroll
    for (int d = 0; d < DEMB; ++d)
      acc += emb[(size_t)tok * DEMB + d] * W1[d * H1 + h];
  }
  xw1[idx] = acc;
}

// ---------------------------------------------------------------------------
// Kernel 2: f32 [K][N] -> bf16 transposed/padded [Nrows][Kp] (K-major for B frags)
// ---------------------------------------------------------------------------
__global__ void prep_bf16_t_kernel(const float* src, unsigned short* dst,
                                   int K, int N, int Kp, int Nrows) {
  size_t idx = (size_t)blockIdx.x * blockDim.x + threadIdx.x;
  if (idx >= (size_t)Nrows * Kp) return;
  int n = (int)(idx / Kp);
  int k = (int)(idx % Kp);
  float v = (k < K && n < N) ? src[(size_t)k * N + n] : 0.f;
  dst[idx] = f2bf(v);
}

// ---------------------------------------------------------------------------
// Kernel 3: fused recurrent core. One workgroup = 16 batch rows, 10 waves.
// Wave w owns output N-tile w. Per step:
//   h1_t = tanh(xw1_t + h1_{t-1} @ U1)          (10 N-tiles x 5 K-steps)
//   h2_t = tanh(h1_t @ W2 + h2_{t-1} @ U2 + b2) (7 N-tiles x (5+4) K-steps)
// State lives in LDS as bf16. All weight B-fragments are hoisted into VGPRs
// before the time loop (14 frags x 8 VGPRs); A-frags for a stage are loaded
// in one batch so a single s_wait_dscnt covers them and the WMMAs issue
// back-to-back.
// ---------------------------------------------------------------------------
__global__ void __launch_bounds__(320) rnn_fused_kernel(
    const float* __restrict__ xw1, const unsigned short* __restrict__ u1t,
    const unsigned short* __restrict__ w2t, const unsigned short* __restrict__ u2t,
    const float* __restrict__ b2, unsigned short* __restrict__ h2g) {
  __shared__ unsigned short h1s[16 * H1P];   // h1 state, bf16
  __shared__ unsigned short h2s[16 * H2KP];  // h2 state, bf16 (K-padded to 128)

  const int tid  = threadIdx.x;
  const int wave = tid >> 5;
  const int lane = tid & 31;
  const int r0   = blockIdx.x * 16;
  const int n0   = wave * 16;          // this wave's N offset
  const int col  = lane & 15;          // C-fragment: N = lane%16
  const int mrow = (lane >> 4) << 3;   // C-fragment: M = vgpr + 8*(lane>=16)

  for (int i = tid; i < 16 * H1P;  i += 320) h1s[i] = 0;   // h_{-1} = 0
  for (int i = tid; i < 16 * H2KP; i += 320) h2s[i] = 0;
  __syncthreads();

  // ---- hoist all loop-invariant weight B-fragments into VGPRs ----
  v16bf bu1[NK1], bw2[NK1], bu2[NK2];
#pragma unroll
  for (int k = 0; k < NK1; ++k) bu1[k] = load_b_frag(u1t, H1P, n0, 32 * k, lane);
  float b2v = 0.f;
  if (wave < 7) {                      // wave-uniform: EXEC stays all-1
#pragma unroll
    for (int k = 0; k < NK1; ++k) bw2[k] = load_b_frag(w2t, H1P, n0, 32 * k, lane);
#pragma unroll
    for (int k = 0; k < NK2; ++k) bu2[k] = load_b_frag(u2t, H2KP, n0, 32 * k, lane);
    int c = n0 + col;
    b2v = (c < H2) ? b2[c] : 0.f;
  }

  for (int t = 0; t < SS; ++t) {
    // ---- stage 1: h1_t = tanh(xw1_t + h1_{t-1} @ U1) ----
    v8f c1;
#pragma unroll
    for (int i = 0; i < 8; ++i)
      c1[i] = xw1[((size_t)(r0 + mrow + i) * SS + t) * H1P + n0 + col];
    {
      v16bf a[NK1];
#pragma unroll
      for (int k = 0; k < NK1; ++k) a[k] = load_a_frag(h1s, H1P, 32 * k, lane);
#pragma unroll
      for (int k = 0; k < NK1; ++k) c1 = WMMA_BF16(a[k], bu1[k], c1);
    }
#pragma unroll
    for (int i = 0; i < 8; ++i) c1[i] = fast_tanh(c1[i]);
    __syncthreads();   // all waves done reading h1_{t-1}
#pragma unroll
    for (int i = 0; i < 8; ++i)
      h1s[(mrow + i) * H1P + n0 + col] = f2bf(c1[i]);
    __syncthreads();   // h1s now holds h1_t

    // ---- stage 2: h2_t (waves 0..6; wave-uniform branch) ----
    v8f c2;
#pragma unroll
    for (int i = 0; i < 8; ++i) c2[i] = b2v;
    if (wave < 7) {
      v16bf a1[NK1], a2[NK2];
#pragma unroll
      for (int k = 0; k < NK1; ++k) a1[k] = load_a_frag(h1s, H1P, 32 * k, lane);
#pragma unroll
      for (int k = 0; k < NK2; ++k) a2[k] = load_a_frag(h2s, H2KP, 32 * k, lane);
#pragma unroll
      for (int k = 0; k < NK1; ++k) c2 = WMMA_BF16(a1[k], bw2[k], c2);
#pragma unroll
      for (int k = 0; k < NK2; ++k) c2 = WMMA_BF16(a2[k], bu2[k], c2);
#pragma unroll
      for (int i = 0; i < 8; ++i) c2[i] = fast_tanh(c2[i]);
    }
    __syncthreads();   // all reads of h2_{t-1} done
    if (wave < 7) {
#pragma unroll
      for (int i = 0; i < 8; ++i)
        h2s[(mrow + i) * H2KP + n0 + col] = f2bf(c2[i]);
    }
    __syncthreads();   // h2s now holds h2_t
  }

  // final h2 (bf16, K padded to 128; pad cols are zero) -> global
  for (int i = tid; i < 16 * H2KP; i += 320)
    h2g[(size_t)(r0 + i / H2KP) * H2KP + (i % H2KP)] = h2s[i];
}

// ---------------------------------------------------------------------------
// Kernel 4: logits = h2 @ Wd + bd, fused row-max. A-frags (h2) stay in VGPRs
// across all N-tiles a wave processes; Wd streamed bf16 with prefetch.
// ---------------------------------------------------------------------------
__global__ void __launch_bounds__(256) dense_logits_kernel(
    const unsigned short* __restrict__ h2g, const unsigned short* __restrict__ wdt,
    const float* __restrict__ bd, float* __restrict__ out,
    unsigned int* __restrict__ rmax) {
  const int wave = threadIdx.x >> 5;
  const int lane = threadIdx.x & 31;
  const int r0   = blockIdx.y * 16;
  const int col  = lane & 15;
  const int mrow = (lane >> 4) << 3;

  v16bf a[NK2];
#pragma unroll
  for (int k = 0; k < NK2; ++k)
    a[k] = load_a_frag(h2g + (size_t)r0 * H2KP, H2KP, 32 * k, lane);

  const int tstride = gridDim.x * 8;
  for (int tile = blockIdx.x * 8 + wave; tile < NT; tile += tstride) {
    const int n0 = tile * 16;
    const int nn = n0 + col;                 // < NDP (padded), guarded vs VV below
    const int tnext = tile + tstride;
    if (tnext < NT)                          // -> global_prefetch_b8
      __builtin_prefetch(wdt + (size_t)(tnext * 16 + col) * H2KP + ((lane < 16) ? 0 : 16), 0, 0);

    const float bdv = (nn < VV) ? bd[nn] : 0.f;
    v8f c;
#pragma unroll
    for (int i = 0; i < 8; ++i) c[i] = bdv;  // bias splat: lane's N is fixed
    {
      v16bf b[NK2];
#pragma unroll
      for (int k = 0; k < NK2; ++k) b[k] = load_b_frag(wdt, H2KP, n0, 32 * k, lane);
#pragma unroll
      for (int k = 0; k < NK2; ++k) c = WMMA_BF16(a[k], b[k], c);
    }

    float mx[8];
#pragma unroll
    for (int i = 0; i < 8; ++i) {
      mx[i] = (nn < VV) ? c[i] : -__builtin_inff();
      if (nn < VV) out[(size_t)(r0 + mrow + i) * VV + nn] = c[i];
    }
    // row-max butterfly across the 16 lanes sharing the same 8 rows (wave32)
#pragma unroll
    for (int m = 1; m < 16; m <<= 1)
#pragma unroll
      for (int i = 0; i < 8; ++i) mx[i] = fmaxf(mx[i], __shfl_xor(mx[i], m, 16));
    if (col == 0) {   // lanes 0 and 16 commit rows r0+mrow .. r0+mrow+7
#pragma unroll
      for (int i = 0; i < 8; ++i)
        atomicMax(rmax + r0 + mrow + i, f32_ordered(mx[i]));
    }
  }
}

// ---------------------------------------------------------------------------
// Kernel 5: e = exp(logit - rowmax), write back, block-reduced atomic row sums
// ---------------------------------------------------------------------------
__global__ void __launch_bounds__(256) exp_sum_kernel(
    float* __restrict__ out, const unsigned int* __restrict__ rmax,
    float* __restrict__ rsum) {
  __shared__ float red[256];
  const int row = blockIdx.y;
  const int c   = blockIdx.x * 256 + threadIdx.x;
  const float m = f32_unordered(rmax[row]);
  float e = 0.f;
  if (c < VV) {
    size_t idx = (size_t)row * VV + c;
    e = __expf(out[idx] - m);
    out[idx] = e;
  }
  red[threadIdx.x] = e;
  __syncthreads();
  for (int s = 128; s > 0; s >>= 1) {
    if (threadIdx.x < s) red[threadIdx.x] += red[threadIdx.x + s];
    __syncthreads();
  }
  if (threadIdx.x == 0) atomicAdd(rsum + row, red[0]);
}

// ---------------------------------------------------------------------------
// Kernel 6: normalize
// ---------------------------------------------------------------------------
__global__ void __launch_bounds__(256) normalize_kernel(
    float* __restrict__ out, const float* __restrict__ rsum) {
  const int row = blockIdx.y;
  const int c   = blockIdx.x * 256 + threadIdx.x;
  if (c < VV) out[(size_t)row * VV + c] *= (1.0f / rsum[row]);
}

// ---------------------------------------------------------------------------
// Launch. Workspace layout (~96 MB):
//   xw1 f32 [B][S][H1P] | U1^T bf16 [160][160] | W2^T bf16 [112][160]
//   U2^T bf16 [112][128] | Wd^T bf16 [50272][128] | h2 bf16 [B][128]
//   rowmax u32 [B] | rowsum f32 [B]
// ---------------------------------------------------------------------------
extern "C" void kernel_launch(void* const* d_in, const int* in_sizes, int n_in,
                              void* d_out, int out_size, void* d_ws, size_t ws_size,
                              hipStream_t stream) {
  const int*   tokens = (const int*)d_in[0];
  const float* emb    = (const float*)d_in[1];
  const float* W1     = (const float*)d_in[2];
  const float* U1     = (const float*)d_in[3];
  const float* b1     = (const float*)d_in[4];
  const float* W2     = (const float*)d_in[5];
  const float* U2     = (const float*)d_in[6];
  const float* b2     = (const float*)d_in[7];
  const float* Wd     = (const float*)d_in[8];
  const float* bd     = (const float*)d_in[9];
  float*       out    = (float*)d_out;

  char* ws = (char*)d_ws;
  constexpr size_t OFF_XW1 = 0;
  constexpr size_t OFF_U1T = OFF_XW1 + (size_t)BB * SS * H1P * 4;
  constexpr size_t OFF_W2T = OFF_U1T + (size_t)H1P * H1P * 2;
  constexpr size_t OFF_U2T = OFF_W2T + (size_t)H2NP * H1P * 2;
  constexpr size_t OFF_WDT = OFF_U2T + (size_t)H2NP * H2KP * 2;
  constexpr size_t OFF_H2G = OFF_WDT + (size_t)NDP * H2KP * 2;
  constexpr size_t OFF_RMX = OFF_H2G + (size_t)BB * H2KP * 2;
  constexpr size_t OFF_RSM = OFF_RMX + (size_t)BB * 4;

  float*          xw1  = (float*)(ws + OFF_XW1);
  unsigned short* u1t  = (unsigned short*)(ws + OFF_U1T);
  unsigned short* w2t  = (unsigned short*)(ws + OFF_W2T);
  unsigned short* u2t  = (unsigned short*)(ws + OFF_U2T);
  unsigned short* wdt  = (unsigned short*)(ws + OFF_WDT);
  unsigned short* h2g  = (unsigned short*)(ws + OFF_H2G);
  unsigned int*   rmax = (unsigned int*)(ws + OFF_RMX);
  float*          rsum = (float*)(ws + OFF_RSM);

  init_stats_kernel<<<(BB + 255) / 256, 256, 0, stream>>>(rmax, rsum);

  {
    size_t n = (size_t)BB * SS * H1P;
    embed_proj_kernel<<<(unsigned)((n + 255) / 256), 256, 0, stream>>>(tokens, emb, W1, b1, xw1);
  }

  prep_bf16_t_kernel<<<(H1P * H1P + 255) / 256, 256, 0, stream>>>(U1, u1t, H1, H1, H1P, H1P);
  prep_bf16_t_kernel<<<(H2NP * H1P + 255) / 256, 256, 0, stream>>>(W2, w2t, H1, H2, H1P, H2NP);
  prep_bf16_t_kernel<<<(H2NP * H2KP + 255) / 256, 256, 0, stream>>>(U2, u2t, H2, H2, H2KP, H2NP);
  {
    size_t n = (size_t)NDP * H2KP;
    prep_bf16_t_kernel<<<(unsigned)((n + 255) / 256), 256, 0, stream>>>(Wd, wdt, H2, VV, H2KP, NDP);
  }

  rnn_fused_kernel<<<BB / 16, 320, 0, stream>>>(xw1, u1t, w2t, u2t, b2, h2g);

  dense_logits_kernel<<<dim3(64, BB / 16), 256, 0, stream>>>(h2g, wdt, bd, out, rmax);

  dim3 sgrid((VV + 255) / 256, BB);
  exp_sum_kernel<<<sgrid, 256, 0, stream>>>(out, rmax, rsum);
  normalize_kernel<<<sgrid, 256, 0, stream>>>(out, rsum);
}
